// GlobalPositionEncDec_89970974917218
// MI455X (gfx1250) — matrix operand order repaired and checked by reading, not compiled, not run on hardware
//
#include <hip/hip_runtime.h>
#include <hip/hip_bf16.h>

// ---------------------------------------------------------------------------
// GlobalPositionEncDec fused kernel for MI455X (gfx1250, wave32, WMMA)
//
//   loc  = PE(coords) @ loc_w + loc_b          (B,64)x(64,128)
//   h    = relu(loc @ fc1_w + fc1_b)           (B,128)x(128,64)
//   pred = h @ fc2_w + fc2_b                   (B,64)x(64,64)
//   loss = mean(-(logsig(pos) + mean_k logsig(-neg_k)))
//
// All three GEMMs run on v_wmma_f32_16x16x32_f16 (f16 in, f32 accumulate).
// ---------------------------------------------------------------------------

typedef __attribute__((ext_vector_type(16))) _Float16 v16h;
typedef __attribute__((ext_vector_type(8)))  _Float16 v8h;
typedef __attribute__((ext_vector_type(8)))  float    v8f;

constexpr int PEDIM = 64;    // sinusoidal feature dim
constexpr int DLOC  = 128;   // location encoder out
constexpr int HID   = 64;    // hidden
constexpr int DIMF  = 64;    // POI feature dim
constexpr int NNEG  = 10;
constexpr int WAVES = 8;     // 256 threads / 32
constexpr int ROWS_PER_BLOCK = WAVES * 16;

// LDS pitches (in elements), padded +8 halves to spread bank accesses
constexpr int WLOC_PITCH = PEDIM + 8;  // 72  (loc_w^T: [128][64])
constexpr int W1_PITCH   = DLOC + 8;   // 136 (fc1_w^T: [64][128])
constexpr int W2_PITCH   = HID + 8;    // 72  (fc2_w^T: [64][64])
constexpr int PE_PITCH   = 72;         // staged PE / H tiles [16][64]
constexpr int LOC_PITCH  = 136;        // staged loc tile [16][128]
constexpr int PRED_PITCH = 72;         // staged pred tile [16][64] (f32)

// LDS region sizes (bytes)
constexpr size_t SZ_WEIGHTS = (size_t)(DLOC * WLOC_PITCH + HID * W1_PITCH + DIMF * W2_PITCH) * 2;
constexpr size_t SZ_BIAS    = (size_t)(DLOC + HID + DIMF) * 4;
constexpr size_t SZ_STAGEA  = (size_t)WAVES * 16 * PE_PITCH * 2;
constexpr size_t SZ_STAGEB  = (size_t)WAVES * 16 * LOC_PITCH * 2;
constexpr size_t SZ_PRED    = (size_t)WAVES * 16 * PRED_PITCH * 4;
constexpr size_t SZ_PART    = (size_t)WAVES * 4;
constexpr size_t SMEM_BYTES = SZ_WEIGHTS + SZ_BIAS + SZ_STAGEA + SZ_STAGEB + SZ_PRED + SZ_PART;

// A fragment (16x32 f16, row = lane&15): halves 0..7 <-> K = k0+hi*8..+7,
// halves 8..15 <-> K = k0+16+hi*8..+7  (ISA 7.12.2, 16-bit A matrix)
__device__ __forceinline__ v16h load_a_frag(const _Float16* A, int pitch, int k0, int lane) {
    const int m = lane & 15, hi = lane >> 4;
    const _Float16* base = A + m * pitch + k0 + hi * 8;
    v8h lo = *(const v8h*)(base);
    v8h hh = *(const v8h*)(base + 16);
    v16h a;
#pragma unroll
    for (int i = 0; i < 8; ++i) { a[i] = lo[i]; a[i + 8] = hh[i]; }
    return a;
}

// B fragment (32x16 f16, col = lane&15): halves 0..7 <-> K = k0+hi*8..+7, halves 8..15 <-> K = k0+16+hi*8..+7.
// Weights are staged K-contiguous (transposed), so this is 2 aligned v8h loads.
__device__ __forceinline__ v16h load_b_frag(const _Float16* Wt, int pitch, int k0, int col0, int lane) {
    const int n = lane & 15, hi = lane >> 4;
    const _Float16* base = Wt + (col0 + n) * pitch + k0 + hi * 8;
    v8h b0 = *(const v8h*)(base);
    v8h b1 = *(const v8h*)(base + 16);
    v16h b;
#pragma unroll
    for (int i = 0; i < 8; ++i) { b[i] = b0[i]; b[i + 8] = b1[i]; }
    return b;
}

__device__ __forceinline__ float logsig(float x) {
    // numerically stable log(sigmoid(x)) = min(x,0) - log1p(exp(-|x|))
    return fminf(x, 0.0f) - __logf(1.0f + __expf(-fabsf(x)));
}

extern "C" __global__ __launch_bounds__(256, 1)
void poi_fused_kernel(const int* __restrict__ pos_t, const float* __restrict__ coords,
                      const int* __restrict__ neg_t, const float* __restrict__ poi_emb,
                      const float* __restrict__ loc_w, const float* __restrict__ loc_b,
                      const float* __restrict__ fc1_w, const float* __restrict__ fc1_b,
                      const float* __restrict__ fc2_w, const float* __restrict__ fc2_b,
                      float* __restrict__ partial) {
    extern __shared__ char smem[];
    _Float16* sWloc = (_Float16*)smem;                               // [128][72]
    _Float16* sW1   = sWloc + DLOC * WLOC_PITCH;                     // [64][136]
    _Float16* sW2   = sW1 + HID * W1_PITCH;                          // [64][72]
    float*    sBloc = (float*)(smem + SZ_WEIGHTS);
    float*    sB1   = sBloc + DLOC;
    float*    sB2   = sB1 + HID;
    _Float16* sStageA = (_Float16*)(smem + SZ_WEIGHTS + SZ_BIAS);    // PE then H
    _Float16* sStageB = sStageA + WAVES * 16 * PE_PITCH;             // loc
    float*    sPred   = (float*)(smem + SZ_WEIGHTS + SZ_BIAS + SZ_STAGEA + SZ_STAGEB);
    float*    sPartial = (float*)(smem + SMEM_BYTES - SZ_PART);

    const int tid  = threadIdx.x;
    const int wave = tid >> 5;
    const int lane = tid & 31;

    // ---- Phase 0: stage f16 weights (K-contiguous / transposed) + biases ----
    for (int i = tid; i < PEDIM * DLOC; i += 256) {
        int k = i / DLOC, n = i % DLOC;
        sWloc[n * WLOC_PITCH + k] = (_Float16)loc_w[i];
    }
    for (int i = tid; i < DLOC * HID; i += 256) {
        int k = i / HID, n = i % HID;
        sW1[n * W1_PITCH + k] = (_Float16)fc1_w[i];
    }
    for (int i = tid; i < HID * DIMF; i += 256) {
        int k = i / DIMF, n = i % DIMF;
        sW2[n * W2_PITCH + k] = (_Float16)fc2_w[i];
    }
    if (tid < DLOC) sBloc[tid] = loc_b[tid];
    if (tid < HID)  sB1[tid]   = fc1_b[tid];
    if (tid < DIMF) sB2[tid]   = fc2_b[tid];
    __syncthreads();

    const int rowBase = blockIdx.x * ROWS_PER_BLOCK + wave * 16;
    _Float16* peA   = sStageA + wave * 16 * PE_PITCH;   // PE tile, later reused for H
    _Float16* locB  = sStageB + wave * 16 * LOC_PITCH;
    float*    predB = sPred   + wave * 16 * PRED_PITCH;

    // ---- Phase 1: sinusoidal position encoding (16 rows x 64 cols, f16) ----
    // col = ci*32 + within; within<16 -> sin(c * 2^within), else cos(c * 2^(within-16))
    {
        const float freq = (float)(1 << (lane & 15));
        const bool isSin = lane < 16;
#pragma unroll
        for (int j = 0; j < 32; ++j) {
            const int r = j >> 1, ci = j & 1;
            const float cv  = coords[(size_t)(rowBase + r) * 2 + ci];
            const float ang = cv * freq;
            const float val = isSin ? __sinf(ang) : __cosf(ang);
            peA[r * PE_PITCH + ci * 32 + lane] = (_Float16)val;
        }
    }
    // same-wave LDS ops are in-order (DScnt) -> no cross-wave sharing, no barrier

    const int nCol = lane & 15, hiHalf = lane >> 4;

    // ---- Phase 2: loc = PE(16x64) @ loc_w(64x128) + b  -> f16 tile ----
#pragma unroll
    for (int t = 0; t < 8; ++t) {
        v8f acc = {};
#pragma unroll
        for (int kk = 0; kk < 2; ++kk) {
            v16h a = load_a_frag(peA, PE_PITCH, kk * 32, lane);
            v16h b = load_b_frag(sWloc, WLOC_PITCH, kk * 32, t * 16, lane);
            acc = __builtin_amdgcn_wmma_f32_16x16x32_f16(false, a, false, b,
                                                         (short)0, acc, false, false);
        }
        const float bias = sBloc[t * 16 + nCol];
#pragma unroll
        for (int r = 0; r < 8; ++r) {
            const int row = r + hiHalf * 8;   // D layout: VGPR r -> M = r + hi*8
            locB[row * LOC_PITCH + t * 16 + nCol] = (_Float16)(acc[r] + bias);
        }
    }

    // ---- Phase 3: h = relu(loc(16x128) @ fc1_w(128x64) + b) -> reuse PE tile ----
    _Float16* hA = peA;
#pragma unroll
    for (int t = 0; t < 4; ++t) {
        v8f acc = {};
#pragma unroll
        for (int kk = 0; kk < 4; ++kk) {
            v16h a = load_a_frag(locB, LOC_PITCH, kk * 32, lane);
            v16h b = load_b_frag(sW1, W1_PITCH, kk * 32, t * 16, lane);
            acc = __builtin_amdgcn_wmma_f32_16x16x32_f16(false, a, false, b,
                                                         (short)0, acc, false, false);
        }
        const float bias = sB1[t * 16 + nCol];
#pragma unroll
        for (int r = 0; r < 8; ++r) {
            const int row = r + hiHalf * 8;
            hA[row * PE_PITCH + t * 16 + nCol] = (_Float16)fmaxf(acc[r] + bias, 0.0f);
        }
    }

    // ---- Phase 4: pred = h(16x64) @ fc2_w(64x64) + b -> f32 tile ----
#pragma unroll
    for (int t = 0; t < 4; ++t) {
        v8f acc = {};
#pragma unroll
        for (int kk = 0; kk < 2; ++kk) {
            v16h a = load_a_frag(hA, PE_PITCH, kk * 32, lane);
            v16h b = load_b_frag(sW2, W2_PITCH, kk * 32, t * 16, lane);
            acc = __builtin_amdgcn_wmma_f32_16x16x32_f16(false, a, false, b,
                                                         (short)0, acc, false, false);
        }
        const float bias = sB2[t * 16 + nCol];
#pragma unroll
        for (int r = 0; r < 8; ++r) {
            const int row = r + hiHalf * 8;
            predB[row * PRED_PITCH + t * 16 + nCol] = acc[r] + bias;
        }
    }

    // ---- Phase 5: NCE-style scoring. Lane pair (l, l^16) shares row = l&15;
    //      each lane handles a 32-dim half, combined via __shfl_xor(16). ----
    {
        const int row  = lane & 15;
        const int hs   = lane >> 4;
        const int gRow = rowBase + row;

        float p[32];
        const float4* pp = (const float4*)(predB + row * PRED_PITCH + hs * 32);
#pragma unroll
        for (int i = 0; i < 8; ++i) {
            float4 v = pp[i];
            p[i * 4 + 0] = v.x; p[i * 4 + 1] = v.y; p[i * 4 + 2] = v.z; p[i * 4 + 3] = v.w;
        }

        auto dot32 = [&](const float* emb) -> float {
            const float4* e = (const float4*)emb;
            float s = 0.0f;
#pragma unroll
            for (int i = 0; i < 8; ++i) {
                float4 v = e[i];
                s = fmaf(p[i * 4 + 0], v.x,
                    fmaf(p[i * 4 + 1], v.y,
                    fmaf(p[i * 4 + 2], v.z,
                    fmaf(p[i * 4 + 3], v.w, s))));
            }
            return s;
        };

        float ps = dot32(poi_emb + (size_t)pos_t[gRow] * DIMF + hs * 32);
        ps += __shfl_xor(ps, 16, 32);

        float negll = 0.0f;
#pragma unroll
        for (int k = 0; k < NNEG; ++k) {
            const int idx = neg_t[(size_t)gRow * NNEG + k];
            float ns = dot32(poi_emb + (size_t)idx * DIMF + hs * 32);
            ns += __shfl_xor(ns, 16, 32);
            negll += logsig(-ns);
        }

        // both half-lanes hold the full row loss -> weight by 0.5
        float acc = -(logsig(ps) + negll * (1.0f / NNEG)) * 0.5f;
#pragma unroll
        for (int off = 16; off; off >>= 1) acc += __shfl_xor(acc, off, 32);
        if (lane == 0) sPartial[wave] = acc;
    }
    __syncthreads();
    if (tid == 0) {
        float s = 0.0f;
#pragma unroll
        for (int w = 0; w < WAVES; ++w) s += sPartial[w];
        partial[blockIdx.x] = s;
    }
}

// Deterministic final reduction (fixed tree, no float atomics)
extern "C" __global__ void poi_reduce_kernel(const float* __restrict__ partial, int n,
                                             float* __restrict__ out, float scale) {
    __shared__ float buf[256];
    const int tid = threadIdx.x;
    float s = 0.0f;
    for (int i = tid; i < n; i += 256) s += partial[i];
    buf[tid] = s;
    __syncthreads();
    for (int off = 128; off; off >>= 1) {
        if (tid < off) buf[tid] += buf[tid + off];
        __syncthreads();
    }
    if (tid == 0) out[0] = buf[0] * scale;
}

extern "C" void kernel_launch(void* const* d_in, const int* in_sizes, int n_in,
                              void* d_out, int out_size, void* d_ws, size_t ws_size,
                              hipStream_t stream) {
    (void)n_in; (void)out_size; (void)ws_size;
    const int*   pos    = (const int*)d_in[0];
    const float* coords = (const float*)d_in[1];
    const int*   negs   = (const int*)d_in[2];
    const float* poi_e  = (const float*)d_in[3];
    const float* loc_w  = (const float*)d_in[4];
    const float* loc_b  = (const float*)d_in[5];
    const float* fc1_w  = (const float*)d_in[6];
    const float* fc1_b  = (const float*)d_in[7];
    const float* fc2_w  = (const float*)d_in[8];
    const float* fc2_b  = (const float*)d_in[9];

    const int nB      = in_sizes[0];                 // 262144
    const int nBlocks = nB / ROWS_PER_BLOCK;         // 2048
    float* partial = (float*)d_ws;

    poi_fused_kernel<<<nBlocks, 256, SMEM_BYTES, stream>>>(
        pos, coords, negs, poi_e, loc_w, loc_b, fc1_w, fc1_b, fc2_w, fc2_b, partial);
    poi_reduce_kernel<<<1, 256, 0, stream>>>(partial, nBlocks, (float*)d_out,
                                             1.0f / (float)nB);
}
